// VolumeRender_1958505087027
// MI455X (gfx1250) — matrix operand order repaired and checked by reading, not compiled, not run on hardware
//
#include <hip/hip_runtime.h>
#include <math.h>

typedef __attribute__((ext_vector_type(16))) _Float16 v16h;
typedef __attribute__((ext_vector_type(8)))  _Float16 v8h;
typedef __attribute__((ext_vector_type(8)))  float    v8f;

#define NVOX 128
#define CH 23
#define CSTRIDE (NVOX * NVOX * NVOX)
#define R_TOTAL 4096
#define P_ALL 128           // 64 coarse + 64 fine
#define N_INIT 256          // N_SAMPLES*4
#define NEAR_F 0.2f
#define FAR_F 6.0f
#define SPAN (FAR_F - NEAR_F) // 5.8
#define INV_NORM (2.0f / 5.12f) // xn = p * INV_NORM - 1  (nxyz = 128*0.04)

// ---------- binary searches on LDS arrays ----------
__device__ __forceinline__ int lb_count_lt(const float* a, int n, float x) {
    int lo = 0, hi = n;
    while (lo < hi) { int m = (lo + hi) >> 1; if (a[m] < x) lo = m + 1; else hi = m; }
    return lo;
}
__device__ __forceinline__ int ub_count_le(const float* a, int n, float x) {
    int lo = 0, hi = n;
    while (lo < hi) { int m = (lo + hi) >> 1; if (a[m] <= x) lo = m + 1; else hi = m; }
    return lo;
}

// ---------- trilinear, align_corners=True, border clamp (SDF volume, 1 channel) ----------
__device__ __forceinline__ float tri_sdf(const float* __restrict__ vol,
                                         float xn, float yn, float zn) {
    float c[3] = {xn, yn, zn};
    int   i0[3];
    float fr[3];
#pragma unroll
    for (int a = 0; a < 3; ++a) {
        float ia = (c[a] + 1.0f) * 0.5f * 127.0f;
        ia = fminf(fmaxf(ia, 0.0f), 127.0f);
        float f0 = floorf(ia);
        i0[a] = (int)f0;
        fr[a] = ia - f0;
    }
    float acc = 0.0f;
#pragma unroll
    for (int k = 0; k < 8; ++k) {
        float w = 1.0f;
        int idx3[3];
#pragma unroll
        for (int a = 0; a < 3; ++a) {
            int off = (k >> a) & 1;
            int ii  = i0[a] + off;
            w *= off ? fr[a] : (1.0f - fr[a]);
            if (ii > 127) ii = 127;   // i0 >= 0 already
            idx3[a] = ii;
        }
        int fo = (idx3[0] * NVOX + idx3[1]) * NVOX + idx3[2];
        acc += w * vol[fo];
    }
    return acc;
}

// ---------- trilinear setup, align_corners=False, zeros padding (attr volume) ----------
__device__ __forceinline__ void tri_attr_setup(float xn, float yn, float zn,
                                               float* w8, int* off8) {
    float c[3] = {xn, yn, zn};
    int   i0[3];
    float fr[3];
#pragma unroll
    for (int a = 0; a < 3; ++a) {
        float ia = ((c[a] + 1.0f) * 128.0f - 1.0f) * 0.5f;
        float f0 = floorf(ia);
        i0[a] = (int)f0;
        fr[a] = ia - f0;
    }
#pragma unroll
    for (int k = 0; k < 8; ++k) {
        float w = 1.0f;
        bool valid = true;
        int idx3[3];
#pragma unroll
        for (int a = 0; a < 3; ++a) {
            int off = (k >> a) & 1;
            int ii  = i0[a] + off;
            w *= off ? fr[a] : (1.0f - fr[a]);
            valid = valid && (ii >= 0) && (ii <= 127);
            ii = ii < 0 ? 0 : (ii > 127 ? 127 : ii);
            idx3[a] = ii;
        }
        w8[k]   = valid ? w : 0.0f;
        off8[k] = (idx3[0] * NVOX + idx3[1]) * NVOX + idx3[2];
    }
}

// One wave32 per ray.
__global__ __launch_bounds__(32) void volrender_kernel(
    const float* __restrict__ rays,      // [R,6]
    const float* __restrict__ vattr,     // [23,128,128,128]
    const float* __restrict__ vsdf,      // [1,128,128,128]
    const float* __restrict__ lnbeta,    // [1]
    float* __restrict__ out)             // rgb|depth|valid|sem flat
{
    __shared__ float    s_cdf[N_INIT];     // cdf[0]=0 .. cdf[255]~=1
    __shared__ float    s_coarse[64];
    __shared__ float    s_fine[64];
    __shared__ float    s_dall[P_ALL];
    __shared__ float    s_sdf[P_ALL];
    __shared__ float    s_buf[32];
    // channel-major staging: s_attr[c][p]; rows 23..31 are zero
    __shared__ __align__(32) _Float16 s_attr[32 * P_ALL];
    __shared__ __align__(32) _Float16 s_tauh[P_ALL];

    const int r    = blockIdx.x;
    const int lane = threadIdx.x;

    // ---- ray setup ----
    float ox = rays[r * 6 + 0], oy = rays[r * 6 + 1], oz = rays[r * 6 + 2];
    float dx = rays[r * 6 + 3], dy = rays[r * 6 + 4], dz = rays[r * 6 + 5];
    dx = (fabsf(dx) < 1e-6f) ? 1e-6f : dx;
    dy = (fabsf(dy) < 1e-6f) ? 1e-6f : dy;
    dz = (fabsf(dz) < 1e-6f) ? 1e-6f : dz;
    float dr = sqrtf(dx * dx + dy * dy + dz * dz);
    float inv_dr = 1.0f / dr;
    dx *= inv_dr; dy *= inv_dr; dz *= inv_dr;

    float beta  = expf(lnbeta[0] * 10.0f);
    float alpha = 1.0f / beta;
    float inv_beta = 1.0f / beta;

    // ================= Phase A: coarse pass (256 samples, 8 per lane) =================
    float a_arr[8];
    float wp_arr[8];
    float prodf = 1.0f;
#pragma unroll
    for (int i = 0; i < 8; ++i) {
        int   s = lane * 8 + i;
        float t = NEAR_F + SPAN * (float)s * (1.0f / 255.0f);
        float px = ox + dx * t, py = oy + dy * t, pz = oz + dz * t;
        float sd = tri_sdf(vsdf, px * INV_NORM - 1.0f, py * INV_NORM - 1.0f,
                           pz * INV_NORM - 1.0f);
        float e     = expf(-fabsf(sd) * inv_beta);
        float sigma = alpha * ((sd >= 0.0f) ? 0.5f * e : 1.0f - 0.5f * e);
        float delta = (s < 255) ? (SPAN / 255.0f) : 1e10f;
        float aa    = 1.0f - expf(-sigma * delta);
        a_arr[i] = aa;
        prodf *= (1.0f - aa + 1e-10f);
    }
    // exclusive cumprod across lanes
    s_buf[lane] = prodf;
    __syncthreads();
    float T = 1.0f;
    for (int j = 0; j < lane; ++j) T *= s_buf[j];
    __syncthreads();
    // weights + (w + 1e-5) for the first 255 samples
    float sum_wp = 0.0f;
#pragma unroll
    for (int i = 0; i < 8; ++i) {
        int   s = lane * 8 + i;
        float w = a_arr[i] * T;
        T *= (1.0f - a_arr[i] + 1e-10f);
        float wp = (s < 255) ? (w + 1e-5f) : 0.0f;
        wp_arr[i] = wp;
        sum_wp += wp;
    }
    s_buf[lane] = sum_wp;
    __syncthreads();
    float pre = 0.0f, tot = 0.0f;
    for (int j = 0; j < 32; ++j) { float v = s_buf[j]; if (j < lane) pre += v; tot += v; }
    __syncthreads();
    float inv_tot = 1.0f / tot;
    float run = pre;
#pragma unroll
    for (int i = 0; i < 8; ++i) {
        int s = lane * 8 + i;
        s_cdf[s] = run * inv_tot;   // cdf[s] = sum_{j<s} pdf_j ; cdf[0]=0
        run += wp_arr[i];
    }
    // coarse bins (64-sample linspace)
    s_coarse[lane]      = NEAR_F + SPAN * (float)lane * (1.0f / 63.0f);
    s_coarse[lane + 32] = NEAR_F + SPAN * (float)(lane + 32) * (1.0f / 63.0f);
    __syncthreads();

    // ================= Phase B: inverse-CDF sampling (64 fine, 2 per lane) ============
#pragma unroll
    for (int ii = 0; ii < 2; ++ii) {
        int   k = lane * 2 + ii;
        float u = ((float)k + 0.5f) * (1.0f / 64.0f);
        int ind   = ub_count_le(s_cdf, N_INIT, u);       // searchsorted right
        int below = ind - 1; below = below < 0 ? 0 : (below > 255 ? 255 : below);
        int above = ind > 255 ? 255 : ind;
        float cg0 = s_cdf[below], cg1 = s_cdf[above];
        float bg0 = NEAR_F + SPAN * (float)below * (1.0f / 255.0f);
        float bg1 = NEAR_F + SPAN * (float)above * (1.0f / 255.0f);
        float den = ((cg1 - cg0) < 1e-5f) ? 1.0f : (cg1 - cg0);
        float tt  = (u - cg0) / den;
        s_fine[k] = bg0 + tt * (bg1 - bg0);
    }
    __syncthreads();

    // ================= Phase C: stable merge of coarse+fine (sorted 128) ==============
#pragma unroll
    for (int ii = 0; ii < 2; ++ii) {
        int   i  = lane * 2 + ii;
        float cv = s_coarse[i];
        s_dall[i + lb_count_lt(s_fine, 64, cv)] = cv;        // coarse before ties
        float fv = s_fine[i];
        s_dall[i + ub_count_le(s_coarse, 64, fv)] = fv;
    }
    __syncthreads();

    // ================= Phase D: fine SDF + 23-channel attr queries ====================
    float sdfsum_local = 0.0f;
    for (int c32 = 0; c32 < 4; ++c32) {
        int   p = c32 * 32 + lane;
        float t = s_dall[p];
        float px = ox + dx * t, py = oy + dy * t, pz = oz + dz * t;
        float xn = px * INV_NORM - 1.0f;
        float yn = py * INV_NORM - 1.0f;
        float zn = pz * INV_NORM - 1.0f;
        float sd = tri_sdf(vsdf, xn, yn, zn);
        s_sdf[p] = sd;
        sdfsum_local += sd;
        float w8[8]; int off8[8];
        tri_attr_setup(xn, yn, zn, w8, off8);
        for (int c = 0; c < CH; ++c) {
            const float* bp = vattr + (size_t)c * CSTRIDE;
            float acc = 0.0f;
#pragma unroll
            for (int k = 0; k < 8; ++k) acc += w8[k] * bp[off8[k]];
            s_attr[c * P_ALL + p] = (_Float16)acc;   // channel-major
        }
#pragma unroll
        for (int c = CH; c < 32; ++c) s_attr[c * P_ALL + p] = (_Float16)0.0f;
    }
    s_buf[lane] = sdfsum_local;
    __syncthreads();
    float sdfsum = 0.0f;
    for (int j = 0; j < 32; ++j) sdfsum += s_buf[j];
    __syncthreads();

    // ================= Phase E: transmittance weights tau over 128 samples ============
    float af[4];
    float prodf2 = 1.0f;
#pragma unroll
    for (int i = 0; i < 4; ++i) {
        int   q  = lane * 4 + i;
        float sd = s_sdf[q];
        float e     = expf(-fabsf(sd) * inv_beta);
        float sigma = alpha * ((sd >= 0.0f) ? 0.5f * e : 1.0f - 0.5f * e);
        float delta = (q < 127) ? (s_dall[q + 1] - s_dall[q]) : 1e10f;
        float aa    = 1.0f - expf(-sigma * delta);
        af[i] = aa;
        prodf2 *= (1.0f - aa + 1e-10f);
    }
    s_buf[lane] = prodf2;
    __syncthreads();
    float T2 = 1.0f;
    for (int j = 0; j < lane; ++j) T2 *= s_buf[j];
    __syncthreads();
    float tsum_l = 0.0f, tdsum_l = 0.0f;
#pragma unroll
    for (int i = 0; i < 4; ++i) {
        int   q   = lane * 4 + i;
        float tau = af[i] * T2;
        T2 *= (1.0f - af[i] + 1e-10f);
        s_tauh[q] = (_Float16)tau;     // staged once as f16 for the WMMA A operand
        tsum_l  += tau;
        tdsum_l += tau * s_dall[q];
    }
    s_buf[lane] = tsum_l;
    __syncthreads();
    float tsum = 0.0f;
    for (int j = 0; j < 32; ++j) tsum += s_buf[j];
    __syncthreads();
    s_buf[lane] = tdsum_l;
    __syncthreads();
    float tdsum = 0.0f;
    for (int j = 0; j < 32; ++j) tdsum += s_buf[j];
    __syncthreads();

    // ================= Phase F: WMMA compositing  D = tau(1x128) x attrs(128x32) ======
    // A (f16 16x32): tau broadcast into ALL 16 rows (branchless, uniform LDS reads);
    //   lane group 0-15 needs K = kb+[0..15]; lanes 16-31 K = kb+16+[0..15].
    // B (f16 32x16): lane = column (channel); lanes 0-15 K 0-15, lanes 16-31 K 16-31.
    //   Channel-major LDS layout makes each lane's 16 halves one contiguous 32B run.
    // Only row M=0 of D (element 0, lanes 0-15) is consumed.
    v8f acc0, acc1;
#pragma unroll
    for (int j = 0; j < 8; ++j) { acc0[j] = 0.0f; acc1[j] = 0.0f; }
    const int n   = lane & 15;
    const int kh  = (lane >> 4) * 16;   // B K-half select
    const int kh2 = (lane >> 4) * 8;    // unused; A also loads with kh
#pragma unroll
    for (int kb = 0; kb < P_ALL; kb += 32) {
        v16h A  = *(const v16h*)(&s_tauh[kb + kh]);        // K kb+kh .. +15
        v16h B0 = *(const v16h*)(&s_attr[n        * P_ALL + kb + kh]);  // ch 0..15
        v16h B1 = *(const v16h*)(&s_attr[(16 + n) * P_ALL + kb + kh]);  // ch 16..31
        acc0 = __builtin_amdgcn_wmma_f32_16x16x32_f16(false, A, false, B0,
                                                      (short)0, acc0, false, false);
        acc1 = __builtin_amdgcn_wmma_f32_16x16x32_f16(false, A, false, B1,
                                                      (short)0, acc1, false, false);
    }

    // ================= Outputs =================
    float dist  = tdsum / (tsum + 1e-10f);
    float depth = dist * inv_dr;

    float* out_rgb   = out;                       // [R,3]
    float* out_depth = out + R_TOTAL * 3;         // [R]
    float* out_valid = out + R_TOTAL * 4;         // [R]
    float* out_sem   = out + R_TOTAL * 5;         // [R,20]

    if (lane < 16) {
        float v0 = acc0[0];   // D[M=0][N=lane]     -> channel lane
        float v1 = acc1[0];   // D[M=0][N=16+lane]  -> channel 16+lane
        int c = lane;
        if (c < 3) out_rgb[r * 3 + c] = v0;
        else       out_sem[r * 20 + (c - 3)] = v0;
        int c2 = 16 + lane;
        if (c2 < CH) out_sem[r * 20 + (c2 - 3)] = v1;
    }
    if (lane == 0) {
        out_depth[r] = depth;
        out_valid[r] = (sdfsum != 128.0f) ? 1.0f : 0.0f;
    }
}

extern "C" void kernel_launch(void* const* d_in, const int* in_sizes, int n_in,
                              void* d_out, int out_size, void* d_ws, size_t ws_size,
                              hipStream_t stream) {
    const float* rays  = (const float*)d_in[0];
    const float* vattr = (const float*)d_in[1];
    const float* vsdf  = (const float*)d_in[2];
    const float* lnb   = (const float*)d_in[3];
    float* out = (float*)d_out;
    volrender_kernel<<<R_TOTAL, 32, 0, stream>>>(rays, vattr, vsdf, lnb, out);
}
